// SparseSelfAttention_5145370821345
// MI455X (gfx1250) — matrix operand order repaired and checked by reading, not compiled, not run on hardware
//
#include <hip/hip_runtime.h>

typedef __attribute__((ext_vector_type(16))) _Float16 v16h;
typedef __attribute__((ext_vector_type(8)))  float    v8f;

namespace {
constexpr int kD    = 1024;   // d_model
constexpr int kH    = 16;     // heads
constexpr int kHD   = 64;     // head dim
constexpr int kSeq  = 4096;   // L
constexpr int kBatch= 4;
constexpr int kTok  = kBatch * kSeq;  // 16384
constexpr int kJp   = 1366;   // atrous seq length: ceil(4098/3)
constexpr int kC2   = 86;     // atrous chunks: ceil(1366/16)
constexpr float kScale = 0.125f; // 64^-0.5
}

union F16Frag { uint4 u[2]; v16h v; };
union Half32  { uint4 u[4]; _Float16 h[32]; };

// A-matrix style fragment: elements 0..7 at p[klo..klo+7], 8..15 at p[16+klo..]
__device__ __forceinline__ v16h frag_split(const _Float16* p, int klo) {
  F16Frag f;
  f.u[0] = *(const uint4*)(p + klo);
  f.u[1] = *(const uint4*)(p + 16 + klo);
  return f.v;
}
// Contiguous-k fragment: 16 contiguous halves p[0..15]
__device__ __forceinline__ v16h frag_contig(const _Float16* p) {
  F16Frag f;
  f.u[0] = *(const uint4*)(p);
  f.u[1] = *(const uint4*)(p + 8);
  return f.v;
}

__global__ void cvt_f32_to_f16(const float* __restrict__ src,
                               _Float16* __restrict__ dst, int n) {
  int i = (blockIdx.x * blockDim.x + threadIdx.x) * 4;
  if (i + 3 < n) {
    float4 f = *(const float4*)(src + i);
    union { _Float16 h[4]; uint2 u; } t;
    t.h[0] = (_Float16)f.x; t.h[1] = (_Float16)f.y;
    t.h[2] = (_Float16)f.z; t.h[3] = (_Float16)f.w;
    *(uint2*)(dst + i) = t.u;
  }
}

// C[m][n] = sum_k A[m][k] * W[n][k] + bias[n]   (A: MxK, W: NxK row-major)
// Block tile 128x128, K staged 32-wide via double-buffered LDS (ping-pong):
// next tile is prefetched into registers while WMMAs consume the current
// buffer -> one barrier per K-step, global latency hidden behind matrix pipe.
// 8 waves, each owns a 32(M) x 64(N) strip: 2 A-frags x 4 B-frags = 8 accums.
template <typename TA, bool OUT_F16>
__global__ __launch_bounds__(256) void gemm_bias(
    const TA* __restrict__ A, const _Float16* __restrict__ W,
    const float* __restrict__ bias, void* __restrict__ Out,
    int M, int N, int K) {
  __shared__ _Float16 As[2][128 * 32];
  __shared__ _Float16 Bs[2][128 * 32];
  const int tid  = threadIdx.x;
  const int wave = tid >> 5;
  const int lane = tid & 31;
  const int ln   = lane & 15;
  const int hi   = lane >> 4;
  const int m0   = blockIdx.y * 128;
  const int n0   = blockIdx.x * 128;
  const int mw   = (wave & 3) * 32;   // wave's M offset inside block tile
  const int nw   = (wave >> 2) * 64;  // wave's N offset inside block tile
  const int srow = tid >> 1;          // staging row 0..127
  const int sseg = (tid & 1) * 16;    // staging half-segment (halves)

  v8f zacc = {};
  v8f acc[2][4];
#pragma unroll
  for (int s = 0; s < 2; ++s)
#pragma unroll
    for (int j = 0; j < 4; ++j) acc[s][j] = zacc;

  const TA*       aptr = A + (size_t)(m0 + srow) * K + sseg;
  const _Float16* bptr = W + (size_t)(n0 + srow) * K + sseg;

  // prefetch registers for one K-tile slice per thread (16 A elems, 16 B halves)
  float4 arf[4];   // used when TA == float
  uint4  arh[2];   // used when TA == _Float16
  uint4  brh[2];

  // ---- prefetch tile kb=0 ----
  if constexpr (sizeof(TA) == 4) {
#pragma unroll
    for (int u = 0; u < 4; ++u) arf[u] = *(const float4*)((const float*)aptr + u * 4);
  } else {
    arh[0] = *(const uint4*)((const _Float16*)aptr);
    arh[1] = *(const uint4*)((const _Float16*)aptr + 8);
  }
  brh[0] = *(const uint4*)(bptr);
  brh[1] = *(const uint4*)(bptr + 8);
  // ---- store to buffer 0 ----
  {
    _Float16* adst = &As[0][srow * 32 + sseg];
    if constexpr (sizeof(TA) == 4) {
#pragma unroll
      for (int u = 0; u < 4; ++u) {
        adst[u * 4 + 0] = (_Float16)arf[u].x; adst[u * 4 + 1] = (_Float16)arf[u].y;
        adst[u * 4 + 2] = (_Float16)arf[u].z; adst[u * 4 + 3] = (_Float16)arf[u].w;
      }
    } else {
      *(uint4*)adst = arh[0]; *(uint4*)(adst + 8) = arh[1];
    }
    _Float16* bdst = &Bs[0][srow * 32 + sseg];
    *(uint4*)bdst = brh[0]; *(uint4*)(bdst + 8) = brh[1];
  }
  __syncthreads();

  const int klo = hi ? 8 : 0;
  const int kbB = hi ? 16 : 0;
  int cur = 0;
  for (int kb = 0; kb < K; kb += 32) {
    const bool hasNext = (kb + 32) < K;
    // issue global prefetch of next tile (overlaps with WMMAs below)
    if (hasNext) {
      if constexpr (sizeof(TA) == 4) {
#pragma unroll
        for (int u = 0; u < 4; ++u)
          arf[u] = *(const float4*)((const float*)aptr + kb + 32 + u * 4);
      } else {
        arh[0] = *(const uint4*)((const _Float16*)aptr + kb + 32);
        arh[1] = *(const uint4*)((const _Float16*)aptr + kb + 32 + 8);
      }
      brh[0] = *(const uint4*)(bptr + kb + 32);
      brh[1] = *(const uint4*)(bptr + kb + 32 + 8);
    }
    // compute on current buffer
    v16h a0 = frag_split(&As[cur][(mw + ln) * 32], klo);
    v16h a1 = frag_split(&As[cur][(mw + 16 + ln) * 32], klo);
#pragma unroll
    for (int j = 0; j < 4; ++j) {
      v16h bf = frag_split(&Bs[cur][(nw + j * 16 + ln) * 32], klo);
      acc[0][j] = __builtin_amdgcn_wmma_f32_16x16x32_f16(false, a0, false, bf,
                                                         (short)0, acc[0][j], false, false);
      acc[1][j] = __builtin_amdgcn_wmma_f32_16x16x32_f16(false, a1, false, bf,
                                                         (short)0, acc[1][j], false, false);
    }
    if (hasNext) {
      const int nxt = cur ^ 1;
      _Float16* adst = &As[nxt][srow * 32 + sseg];
      if constexpr (sizeof(TA) == 4) {
#pragma unroll
        for (int u = 0; u < 4; ++u) {
          adst[u * 4 + 0] = (_Float16)arf[u].x; adst[u * 4 + 1] = (_Float16)arf[u].y;
          adst[u * 4 + 2] = (_Float16)arf[u].z; adst[u * 4 + 3] = (_Float16)arf[u].w;
        }
      } else {
        *(uint4*)adst = arh[0]; *(uint4*)(adst + 8) = arh[1];
      }
      _Float16* bdst = &Bs[nxt][srow * 32 + sseg];
      *(uint4*)bdst = brh[0]; *(uint4*)(bdst + 8) = brh[1];
      __syncthreads();
      cur = nxt;
    }
  }

#pragma unroll
  for (int s = 0; s < 2; ++s) {
#pragma unroll
    for (int j = 0; j < 4; ++j) {
      int n = n0 + nw + j * 16 + ln;
      float bval = bias[n];
#pragma unroll
      for (int r = 0; r < 8; ++r) {
        int m = m0 + mw + s * 16 + r + hi * 8;
        float v = acc[s][j][r] + bval;
        if (OUT_F16) ((_Float16*)Out)[(size_t)m * N + n] = (_Float16)v;
        else         ((float*)Out)[(size_t)m * N + n] = v;
      }
    }
  }
}

// One wave = one (chunk, head). Computes S^T = K*Q^T via WMMA so that the
// softmaxed probabilities land directly in the A-fragment layout for O = P*V.
// mode atrous=0: write Sh; atrous=1: Sh += (stream-ordered, no races).
__global__ __launch_bounds__(256) void attn_chunk(
    const _Float16* __restrict__ Qh, const _Float16* __restrict__ Kh,
    const _Float16* __restrict__ Vh,
    const float* __restrict__ bq, const float* __restrict__ bk,
    const float* __restrict__ bv,
    _Float16* __restrict__ Sh, int atrous, int total) {
  __shared__ _Float16 smem[8 * 3072];
  const int wave = threadIdx.x >> 5;
  const int lane = threadIdx.x & 31;
  const int ln   = lane & 15;
  const int hi   = lane >> 4;
  const int idx  = blockIdx.x * 8 + wave;
  const bool active = idx < total;

  _Float16* Qs = &smem[wave * 3072];       // [row 0..15][k 0..63]
  _Float16* Ks = Qs + 1024;                // [row 0..15][k 0..63]
  _Float16* Vt = Qs + 2048;                // transposed: [d 0..63][row 0..15]

  int b = 0, c = 0, h = 0, dd = 0;
  if (atrous) {
    int s = idx / (kC2 * kH);
    int rem = idx % (kC2 * kH);
    c = rem / kH; h = rem % kH;
    b = s / 3; dd = s % 3;
  } else {
    b = idx / (256 * kH);
    int rem = idx % (256 * kH);
    c = rem / kH; h = rem % kH;
  }

  if (active) {
    const int i   = ln;        // tile row this lane stages
    const int off = hi * 32;   // which half of the 64-dim head vector
    bool rv; long grow = 0;
    if (atrous) {
      int j = c * 16 + i;
      int l = j * 3 + dd;
      rv = (j < kJp) && (l < kSeq);
      grow = (long)b * kSeq + l;
    } else {
      rv = true;
      grow = (long)b * kSeq + c * 16 + i;
    }
    Half32 tq, tk, tv;
    if (rv) {
      size_t go = (size_t)grow * kD + h * kHD + off;
      const uint4* pq = (const uint4*)(Qh + go);
      const uint4* pk = (const uint4*)(Kh + go);
      const uint4* pv = (const uint4*)(Vh + go);
#pragma unroll
      for (int u = 0; u < 4; ++u) { tq.u[u] = pq[u]; tk.u[u] = pk[u]; tv.u[u] = pv[u]; }
    } else {  // pad token: q/k/v == bias
      int bo = h * kHD + off;
#pragma unroll
      for (int u = 0; u < 32; ++u) {
        tq.h[u] = (_Float16)bq[bo + u];
        tk.h[u] = (_Float16)bk[bo + u];
        tv.h[u] = (_Float16)bv[bo + u];
      }
    }
#pragma unroll
    for (int u = 0; u < 4; ++u) {
      *(uint4*)&Qs[i * 64 + off + u * 8] = tq.u[u];
      *(uint4*)&Ks[i * 64 + off + u * 8] = tk.u[u];
    }
#pragma unroll
    for (int u = 0; u < 32; ++u) Vt[(off + u) * 16 + i] = tv.h[u];
  }
  __syncthreads();
  if (!active) return;

  // S^T = K * Q^T : A = K-tile (M=key), B[k][n] = Q[n][k] (N=query)
  v8f c0 = {};
  const int klo = hi ? 8 : 0;
  const int kbB = hi ? 16 : 0;
#pragma unroll
  for (int kb = 0; kb < 64; kb += 32) {
    v16h a  = frag_split(&Ks[ln * 64 + kb], klo);
    v16h bf = frag_split(&Qs[ln * 64 + kb], klo);
    c0 = __builtin_amdgcn_wmma_f32_16x16x32_f16(false, a, false, bf,
                                                (short)0, c0, false, false);
  }
  // lane holds S[query=ln][key = r + hi*8]; softmax over keys
  float sc[8]; float mx = -3.0e38f;
#pragma unroll
  for (int r = 0; r < 8; ++r) { sc[r] = c0[r] * kScale; mx = fmaxf(mx, sc[r]); }
  mx = fmaxf(mx, __shfl_xor(mx, 16, 32));
  float sum = 0.f;
#pragma unroll
  for (int r = 0; r < 8; ++r) { sc[r] = __expf(sc[r] - mx); sum += sc[r]; }
  sum += __shfl_xor(sum, 16, 32);
  float inv = 1.0f / sum;

  // P already matches A-fragment layout (K padded 16->32 with zeros)
  v16h ap;
#pragma unroll
  for (int e = 0; e < 16; ++e) ap[e] = (_Float16)0.f;
#pragma unroll
  for (int r = 0; r < 8; ++r) ap[r] = (_Float16)(sc[r] * inv);

#pragma unroll
  for (int nt = 0; nt < 4; ++nt) {
    v16h bvf;
    {
      F16Frag f;
      f.u[0] = *(const uint4*)&Vt[(nt * 16 + ln) * 16 + hi * 8];  // B[k][n] = V[k][nt*16+n], k = 8*hi + e
      f.u[1] = make_uint4(0u, 0u, 0u, 0u);  // padded K half
      bvf = f.v;
    }
    v8f o = {};
    o = __builtin_amdgcn_wmma_f32_16x16x32_f16(false, ap, false, bvf,
                                               (short)0, o, false, false);
#pragma unroll
    for (int r = 0; r < 8; ++r) {
      int m = r + hi * 8;  // query row
      bool rv; long grow = 0;
      if (atrous) {
        int j = c * 16 + m;
        int l = j * 3 + dd;
        rv = (j < kJp) && (l < kSeq);
        grow = (long)b * kSeq + l;
      } else {
        rv = true;
        grow = (long)b * kSeq + c * 16 + m;
      }
      if (rv) {
        size_t oi = (size_t)grow * kD + h * kHD + nt * 16 + ln;
        if (atrous) Sh[oi] = (_Float16)((float)Sh[oi] + o[r]);
        else        Sh[oi] = (_Float16)o[r];
      }
    }
  }
}

extern "C" void kernel_launch(void* const* d_in, const int* in_sizes, int n_in,
                              void* d_out, int out_size, void* d_ws, size_t ws_size,
                              hipStream_t stream) {
  (void)in_sizes; (void)n_in; (void)out_size; (void)ws_size;
  const float* x  = (const float*)d_in[0];
  const float* Wq = (const float*)d_in[1];
  const float* bq = (const float*)d_in[2];
  const float* Wk = (const float*)d_in[3];
  const float* bk = (const float*)d_in[4];
  const float* Wv = (const float*)d_in[5];
  const float* bv = (const float*)d_in[6];
  const float* Wo = (const float*)d_in[7];
  const float* bo = (const float*)d_in[8];
  float* out = (float*)d_out;

  char* ws = (char*)d_ws;
  const size_t MB = 1024ull * 1024ull;
  _Float16* Wqh = (_Float16*)(ws + 0 * MB);
  _Float16* Wkh = (_Float16*)(ws + 2 * MB);
  _Float16* Wvh = (_Float16*)(ws + 4 * MB);
  _Float16* Woh = (_Float16*)(ws + 6 * MB);
  _Float16* Qh  = (_Float16*)(ws + 8 * MB);
  _Float16* Kh  = (_Float16*)(ws + 40 * MB);
  _Float16* Vh  = (_Float16*)(ws + 72 * MB);
  _Float16* Sh  = (_Float16*)(ws + 104 * MB);

  const int wn = kD * kD;  // 1M weights per matrix
  dim3 cblk(256), cgrd(wn / (4 * 256));
  cvt_f32_to_f16<<<cgrd, cblk, 0, stream>>>(Wq, Wqh, wn);
  cvt_f32_to_f16<<<cgrd, cblk, 0, stream>>>(Wk, Wkh, wn);
  cvt_f32_to_f16<<<cgrd, cblk, 0, stream>>>(Wv, Wvh, wn);
  cvt_f32_to_f16<<<cgrd, cblk, 0, stream>>>(Wo, Woh, wn);

  dim3 gblk(256), ggrd(kD / 128, kTok / 128);  // (8, 128)
  gemm_bias<float, true><<<ggrd, gblk, 0, stream>>>(x, Wqh, bq, Qh, kTok, kD, kD);
  gemm_bias<float, true><<<ggrd, gblk, 0, stream>>>(x, Wkh, bk, Kh, kTok, kD, kD);
  gemm_bias<float, true><<<ggrd, gblk, 0, stream>>>(x, Wvh, bv, Vh, kTok, kD, kD);

  const int nLocal  = kBatch * 256 * kH;          // 16384
  const int nAtrous = kBatch * 3 * kC2 * kH;      // 16512
  attn_chunk<<<dim3(nLocal / 8),  256, 0, stream>>>(Qh, Kh, Vh, bq, bk, bv, Sh, 0, nLocal);
  attn_chunk<<<dim3(nAtrous / 8), 256, 0, stream>>>(Qh, Kh, Vh, bq, bk, bv, Sh, 1, nAtrous);

  gemm_bias<_Float16, false><<<ggrd, gblk, 0, stream>>>(Sh, Woh, bo, out, kTok, kD, kD);
}
